// FullQDisentangledVAE_53171695124626
// MI455X (gfx1250) — matrix operand order repaired and checked by reading, not compiled, not run on hardware
//
#include <hip/hip_runtime.h>
#include <hip/hip_bf16.h>

// ---------------------------------------------------------------------------
// Types for CDNA5 WMMA
// ---------------------------------------------------------------------------
typedef __attribute__((ext_vector_type(16))) __bf16 v16bf;
typedef __attribute__((ext_vector_type(8)))  float  v8f;

#define B_  64
#define T_  128
#define H_  1024
#define HH_ 512
#define Z_  256
#define BLK_ 4

__device__ __forceinline__ __bf16 f2bf(float f) {
  unsigned u = __builtin_bit_cast(unsigned, f);
  unsigned r = u + 0x7FFFu + ((u >> 16) & 1u);   // round-to-nearest-even
  unsigned short s = (unsigned short)(r >> 16);
  return __builtin_bit_cast(__bf16, s);
}
__device__ __forceinline__ float sigf(float x) { return 1.0f / (1.0f + expf(-x)); }

// ---------------------------------------------------------------------------
// Generic bf16 WMMA GEMM:  C[M,N] = A_f32[M,K] * Wt_bf16[N,K]^T (+bias) (+Cin)
// grid = (N/64, M/64), block = 128 (4 waves). Wave w -> rows m0..m0+15,
// 4 N-subtiles of 16. B tile (64 x 32 bf16, 4KB) is staged in LDS once per
// K-step (shared by the 4 waves, 4x less L2 traffic), double-buffered so the
// global fetch of tile i+1 overlaps the v_wmma_f32_16x16x32_bf16 on tile i.
// All 4 B fragments are pulled from LDS before the WMMA chain so ds latency
// hides under the matrix pipe (progressive s_wait_dscnt instead of 0x0).
// ---------------------------------------------------------------------------
__global__ __launch_bounds__(128)
void gemm_bf16_wmma(const float* __restrict__ A, long lda,
                    const __bf16* __restrict__ Wt, long ldwt,
                    const float* __restrict__ bias,
                    const float* __restrict__ Cin, long ldcin,
                    float* __restrict__ C, long ldc, int K) {
  __shared__ __bf16 Bs[2][64 * 32];   // [buf][n(0..63) * 32 + k(0..31)]
  const int tid  = threadIdx.x;
  const int lane = tid & 31;
  const int wave = tid >> 5;
  const int hi   = lane >> 4;     // half-wave select
  const int ml   = lane & 15;
  const long m0  = ((long)blockIdx.y * 4 + wave) * 16;
  const long n0  = (long)blockIdx.x * 64;

  v8f acc[4];
#pragma unroll
  for (int i = 0; i < 4; ++i)
#pragma unroll
    for (int j = 0; j < 8; ++j) acc[i][j] = 0.0f;

  // cooperative B-tile staging: 128 threads x 16 bf16 (32B) = 64 rows x 32 k
  const int  rrow  = tid >> 1;
  const int  rhalf = (tid & 1) * 16;
  const __bf16* wsrc = Wt + (n0 + rrow) * ldwt + rhalf;

  const int nk = K >> 5;
  __bf16 stage[16];
#pragma unroll
  for (int e = 0; e < 16; ++e) stage[e] = wsrc[e];
  {
    __bf16* d = &Bs[0][rrow * 32 + rhalf];
#pragma unroll
    for (int e = 0; e < 16; ++e) d[e] = stage[e];
  }

  const float* ap = A + (m0 + ml) * lda + hi * 16;

  for (int i = 0; i < nk; ++i) {
    const int cur = i & 1;
    __syncthreads();                       // Bs[cur] ready; Bs[cur^1] free
    if (i + 1 < nk) {                      // issue global fetch of next tile
      const __bf16* s = wsrc + (long)(i + 1) * 32;
#pragma unroll
      for (int e = 0; e < 16; ++e) stage[e] = s[e];
    }
    const int k0 = i << 5;
    // A fragment: contiguous K order (lanes 0-15: K 0..15; 16-31: K 16..31)
    v16bf a;
#pragma unroll
    for (int e = 0; e < 8; ++e) a[e]     = f2bf(ap[k0 + e]);
#pragma unroll
    for (int e = 0; e < 8; ++e) a[8 + e] = f2bf(ap[k0 + 8 + e]);
    __builtin_prefetch(ap + k0 + 32, 0, 1);
    // pull ALL 4 B fragments from LDS first (clause the ds_loads), then issue
    // the 4 WMMAs back-to-back so LDS latency overlaps the matrix pipe.
    v16bf b[4];
#pragma unroll
    for (int nt = 0; nt < 4; ++nt) {
      const __bf16* bp = &Bs[cur][(nt * 16 + ml) * 32 + hi * 16];
#pragma unroll
      for (int e = 0; e < 16; ++e) b[nt][e] = bp[e];
    }
#pragma unroll
    for (int nt = 0; nt < 4; ++nt) {
      acc[nt] = __builtin_amdgcn_wmma_f32_16x16x32_bf16(
          false, a, false, b[nt], (short)0, acc[nt], false, false);
    }
    if (i + 1 < nk) {                      // store next tile (safe: barrier above)
      __bf16* d = &Bs[cur ^ 1][rrow * 32 + rhalf];
#pragma unroll
      for (int e = 0; e < 16; ++e) d[e] = stage[e];
    }
  }
#pragma unroll
  for (int nt = 0; nt < 4; ++nt) {
    const long n = n0 + nt * 16 + ml;
    const float bv = bias ? bias[n] : 0.0f;
#pragma unroll
    for (int v = 0; v < 8; ++v) {         // C VGPR v: lanes 0-15 M=v, 16-31 M=v+8
      const long m = m0 + v + hi * 8;
      float val = acc[nt][v] + bv;
      if (Cin) val += Cin[m * ldcin + n];
      C[m * ldc + n] = val;
    }
  }
}

// ---------------------------------------------------------------------------
// Weight convert + transpose: src f32 [K,N] -> dst bf16 [N,K]
// ---------------------------------------------------------------------------
__global__ void cvtT_kernel(const float* __restrict__ src, __bf16* __restrict__ dst,
                            int K, int N) {
  long idx = (long)blockIdx.x * blockDim.x + threadIdx.x;
  if (idx >= (long)K * N) return;
  int k = (int)(idx / N), n = (int)(idx % N);
  dst[(long)n * K + k] = f2bf(src[(long)k * N + n]);
}

__global__ void zero_kernel(float* __restrict__ p, long n) {
  long i = (long)blockIdx.x * blockDim.x + threadIdx.x;
  if (i < n) p[i] = 0.0f;
}
__global__ void initwt_kernel(float* __restrict__ wt, float* __restrict__ outwt) {
  int i = threadIdx.x;            // 256 threads: wt0 = ones(64,4); wt_all[0] = wt0
  wt[i] = 1.0f; outwt[i] = 1.0f;
}

// ---------------------------------------------------------------------------
// LSTM gate: G[64,2048] (xg + h@Wh, biased), c[64,512] updated in place,
// h written to hout[:, tcol, :] (stride ldo).
// ---------------------------------------------------------------------------
__global__ void lstm_gate_kernel(const float* __restrict__ G, float* __restrict__ c,
                                 float* __restrict__ hout, long ldo) {
  int idx = blockIdx.x * blockDim.x + threadIdx.x;   // 64*512
  int bb = idx >> 9, j = idx & 511;
  const float* g = G + (long)bb * 2048;
  float i = sigf(g[j]);
  float f = sigf(g[512 + j]);
  float gg = tanhf(g[1024 + j]);
  float o = sigf(g[1536 + j]);
  float cn = f * c[idx] + i * gg;
  c[idx] = cn;
  hout[(long)bb * ldo + j] = o * tanhf(cn);
}

// ---------------------------------------------------------------------------
// GRU gate: xg row (ldx) + Gh[64,3072]; h_new -> hout
// ---------------------------------------------------------------------------
__global__ void gru_gate_kernel(const float* __restrict__ xg, long ldx,
                                const float* __restrict__ Gh,
                                const float* __restrict__ hprev, long ldh,
                                float* __restrict__ hout, long ldo) {
  int idx = blockIdx.x * blockDim.x + threadIdx.x;   // 64*1024
  int bb = idx >> 10, j = idx & 1023;
  const float* xr = xg + (long)bb * ldx;
  const float* gh = Gh + (long)bb * 3072;
  float r = sigf(xr[j] + gh[j]);
  float z = sigf(xr[1024 + j] + gh[1024 + j]);
  float n = tanhf(xr[2048 + j] + r * gh[2048 + j]);
  float hp = hprev[(long)bb * ldh + j];
  hout[(long)bb * ldo + j] = (1.0f - z) * n + z * hp;
}

// ---------------------------------------------------------------------------
// Block-GRU cells (4 blocks), weighted update, mean -> z_all & curr_layer out
// ---------------------------------------------------------------------------
__global__ void blk_gate_kernel(const float* __restrict__ Gx, const float* __restrict__ Gh,
                                float* __restrict__ hblk, const float* __restrict__ wt,
                                float* __restrict__ zall, float* __restrict__ outcurr,
                                int t) {
  int idx = blockIdx.x * blockDim.x + threadIdx.x;   // 64*1024
  int bb = idx >> 10, j = idx & 1023;
  float s = 0.0f;
#pragma unroll
  for (int b = 0; b < BLK_; ++b) {
    const float* gx = Gx + ((long)b * 64 + bb) * 3072;
    const float* gh = Gh + ((long)b * 64 + bb) * 3072;
    float r = sigf(gx[j] + gh[j]);
    float z = sigf(gx[1024 + j] + gh[1024 + j]);
    float n = tanhf(gx[2048 + j] + r * gh[2048 + j]);
    long hix = (long)b * 65536 + idx;
    float hp = hblk[hix];
    float hg = (1.0f - z) * n + z * hp;
    float w = wt[bb * 4 + b];
    float hn = w * hg + (1.0f - w) * hp;
    hblk[hix] = hn;
    s += hn;
  }
  float za = s * 0.25f;
  zall[idx] = za;
  outcurr[((long)t * 64 + bb) * 1024 + j] = za;
}

// z_all@Ww + bw -> softmax -> cumsum -> wt & wt_all[t+1]
__global__ void wt_softmax_kernel(const float* __restrict__ zall,
                                  const float* __restrict__ Ww, const float* __restrict__ bw,
                                  float* __restrict__ wt, float* __restrict__ outwt, int t) {
  int bb = blockIdx.x, tid = threadIdx.x;            // 64 blocks x 128 threads
  float p0 = 0, p1 = 0, p2 = 0, p3 = 0;
  for (int j = tid; j < 1024; j += 128) {
    float v = zall[bb * 1024 + j];
    p0 += v * Ww[j * 4 + 0]; p1 += v * Ww[j * 4 + 1];
    p2 += v * Ww[j * 4 + 2]; p3 += v * Ww[j * 4 + 3];
  }
  __shared__ float s[4][128];
  s[0][tid] = p0; s[1][tid] = p1; s[2][tid] = p2; s[3][tid] = p3;
  __syncthreads();
  if (tid == 0) {
    float l[4];
#pragma unroll
    for (int c = 0; c < 4; ++c) {
      float acc = bw[c];
      for (int i = 0; i < 128; ++i) acc += s[c][i];
      l[c] = acc;
    }
    float mx = fmaxf(fmaxf(l[0], l[1]), fmaxf(l[2], l[3]));
    float e[4], se = 0;
#pragma unroll
    for (int c = 0; c < 4; ++c) { e[c] = expf(l[c] - mx); se += e[c]; }
    float cum = 0;
#pragma unroll
    for (int c = 0; c < 4; ++c) {
      cum += e[c] / se;
      wt[bb * 4 + c] = cum;
      outwt[(long)(t + 1) * 256 + bb * 4 + c] = cum;
    }
  }
}

// posterior split + reparameterization
__global__ void post_split_kernel(const float* __restrict__ P, const float* __restrict__ eps,
                                  float* __restrict__ om, float* __restrict__ os,
                                  float* __restrict__ oz) {
  long idx = (long)blockIdx.x * blockDim.x + threadIdx.x;  // B*T*Z
  if (idx >= (long)B_ * T_ * Z_) return;
  long row = idx / Z_; int z = (int)(idx % Z_);
  float m = P[row * 512 + z];
  float sc = sigf(P[row * 512 + 256 + z]);
  om[idx] = m; os[idx] = sc; oz[idx] = m + eps[idx] * sc;
}

__global__ void prior_split_kernel(const float* __restrict__ Pc, const float* __restrict__ P0,
                                   float* __restrict__ om, float* __restrict__ os) {
  long idx = (long)blockIdx.x * blockDim.x + threadIdx.x;  // B*T*Z
  if (idx >= (long)B_ * T_ * Z_) return;
  long row = idx / Z_; int z = (int)(idx % Z_);
  int t = (int)(row % T_); int bb = (int)(row / T_);
  const float* p = (t == 0) ? (P0 + (long)bb * 512)
                            : (Pc + ((long)(t - 1) * 64 + bb) * 512);
  om[idx] = p[z]; os[idx] = sigf(p[256 + z]);
}

__global__ void copy_last_kernel(const float* __restrict__ gru, float* __restrict__ out) {
  long idx = (long)blockIdx.x * blockDim.x + threadIdx.x;  // 64*127*1024
  if (idx >= (long)B_ * (T_ - 1) * H_) return;
  long bb = idx / ((long)(T_ - 1) * H_);
  long r  = idx % ((long)(T_ - 1) * H_);
  out[idx] = gru[bb * ((long)T_ * H_) + r];
}

// ---------------------------------------------------------------------------
extern "C" void kernel_launch(void* const* d_in, const int* in_sizes, int n_in,
                              void* d_out, int out_size, void* d_ws, size_t ws_size,
                              hipStream_t stream) {
  const float* x    = (const float*)d_in[0];
  const float* eps  = (const float*)d_in[1];
  const float* Wx_f = (const float*)d_in[2];
  const float* Wh_f = (const float*)d_in[3];
  const float* b_f  = (const float*)d_in[4];
  const float* Wx_b = (const float*)d_in[5];
  const float* Wh_b = (const float*)d_in[6];
  const float* b_b  = (const float*)d_in[7];
  const float* Wx_g = (const float*)d_in[8];
  const float* Wh_g = (const float*)d_in[9];
  const float* bx_g = (const float*)d_in[10];
  const float* bh_g = (const float*)d_in[11];
  const float* Wp   = (const float*)d_in[12];
  const float* bp   = (const float*)d_in[13];
  const float* Wpr  = (const float*)d_in[14];
  const float* bpr  = (const float*)d_in[15];
  const float* Ww   = (const float*)d_in[16];
  const float* bw   = (const float*)d_in[17];
  const float* Wx_c = (const float*)d_in[18];
  const float* Wh_c = (const float*)d_in[19];
  const float* bx_c = (const float*)d_in[20];
  const float* bh_c = (const float*)d_in[21];
  (void)in_sizes; (void)n_in; (void)out_size; (void)ws_size;

  float* out = (float*)d_out;
  float* o_pm   = out;                 // (B,T,Z)
  float* o_ps   = out + 2097152;       // (B,T,Z)
  float* o_prm  = out + 4194304;       // (B,T,Z)
  float* o_prs  = out + 6291456;       // (B,T,Z)
  float* o_z    = out + 8388608;       // (B,T,Z)
  float* o_wt   = out + 10485760;      // (T,B,BLK)
  float* o_curr = out + 10518528;      // (127,B,H)
  float* o_last = out + 18841600;      // (B,127,H)

  // --- workspace bump allocator -------------------------------------------
  char* wsb = (char*)d_ws; size_t off = 0;
  auto alloc = [&](size_t bytes) -> char* {
    char* p = wsb + off; off = (off + bytes + 255) & ~(size_t)255; return p;
  };
  __bf16* WxfT = (__bf16*)alloc(2048L * 1024 * 2);
  __bf16* WhfT = (__bf16*)alloc(2048L * 512 * 2);
  __bf16* WxbT = (__bf16*)alloc(2048L * 1024 * 2);
  __bf16* WhbT = (__bf16*)alloc(2048L * 512 * 2);
  __bf16* WxgT = (__bf16*)alloc(3072L * 1024 * 2);
  __bf16* WhgT = (__bf16*)alloc(3072L * 1024 * 2);
  __bf16* WpT  = (__bf16*)alloc(512L * 1024 * 2);
  __bf16* WprT = (__bf16*)alloc(512L * 1024 * 2);
  __bf16* WxcT = (__bf16*)alloc(4L * 3072 * 1024 * 2);
  __bf16* WhcT = (__bf16*)alloc(4L * 3072 * 1024 * 2);
  float* R1   = (float*)alloc(8192L * 3072 * 4);  // xg_f -> xg_b -> xg_g (reused)
  float* fwd  = (float*)alloc(8192L * 512 * 4);   // later reused as Ppost
  float* bwd  = (float*)alloc(8192L * 512 * 4);   // later reused as Pc
  float* gru  = (float*)alloc(8192L * 1024 * 4);
  float* Gtmp = (float*)alloc(64L * 3072 * 4);
  float* Gx   = (float*)alloc(4L * 64 * 3072 * 4);
  float* Ghb  = (float*)alloc(4L * 64 * 3072 * 4);
  float* hblk = (float*)alloc(4L * 64 * 1024 * 4);
  float* zbuf = (float*)alloc(64L * 1024 * 4);
  float* c_f  = (float*)alloc(64L * 512 * 4);
  float* c_b  = (float*)alloc(64L * 512 * 4);
  float* zall = (float*)alloc(64L * 1024 * 4);
  float* wtb  = (float*)alloc(256L * 4);
  float* P0   = (float*)alloc(64L * 512 * 4);
  float* Ppost = fwd;
  float* Pc    = bwd;

  auto cvt = [&](const float* s, __bf16* d, int K, int N) {
    long n = (long)K * N;
    cvtT_kernel<<<dim3((unsigned)((n + 255) / 256)), dim3(256), 0, stream>>>(s, d, K, N);
  };
  auto gemm = [&](const float* A, long lda, const __bf16* Wt, long ldwt,
                  const float* bias, const float* Cin, long ldcin,
                  float* C, long ldc, int M, int N, int K) {
    gemm_bf16_wmma<<<dim3((unsigned)(N / 64), (unsigned)(M / 64)), dim3(128), 0, stream>>>(
        A, lda, Wt, ldwt, bias, Cin, ldcin, C, ldc, K);
  };
  auto zero = [&](float* p, long n) {
    zero_kernel<<<dim3((unsigned)((n + 255) / 256)), dim3(256), 0, stream>>>(p, n);
  };

  // --- weight conversion (f32 -> transposed bf16), once per call ----------
  cvt(Wx_f, WxfT, 1024, 2048); cvt(Wh_f, WhfT, 512, 2048);
  cvt(Wx_b, WxbT, 1024, 2048); cvt(Wh_b, WhbT, 512, 2048);
  cvt(Wx_g, WxgT, 1024, 3072); cvt(Wh_g, WhgT, 1024, 3072);
  cvt(Wp, WpT, 1024, 512);     cvt(Wpr, WprT, 1024, 512);
  for (int b = 0; b < BLK_; ++b) {
    cvt(Wx_c + (long)b * 1024 * 3072, WxcT + (long)b * 3072 * 1024, 1024, 3072);
    cvt(Wh_c + (long)b * 1024 * 3072, WhcT + (long)b * 3072 * 1024, 1024, 3072);
  }
  zero(c_f, 32768); zero(c_b, 32768); zero(hblk, 262144); zero(zbuf, 65536);
  initwt_kernel<<<dim3(1), dim3(256), 0, stream>>>(wtb, o_wt);

  // --- LSTM forward --------------------------------------------------------
  gemm(x, 1024, WxfT, 1024, b_f, nullptr, 0, R1, 2048, 8192, 2048, 1024);  // xg_f
  for (int t = 0; t < T_; ++t) {
    const float* Ap = (t == 0) ? zbuf : (fwd + (long)(t - 1) * HH_);
    long lda = (t == 0) ? 512 : (long)T_ * HH_;
    gemm(Ap, lda, WhfT, 512, nullptr, R1 + (long)t * 2048, (long)T_ * 2048,
         Gtmp, 2048, 64, 2048, 512);
    lstm_gate_kernel<<<dim3(128), dim3(256), 0, stream>>>(
        Gtmp, c_f, fwd + (long)t * HH_, (long)T_ * HH_);
  }
  // --- LSTM backward (natural-order xg, reversed scan) ---------------------
  gemm(x, 1024, WxbT, 1024, b_b, nullptr, 0, R1, 2048, 8192, 2048, 1024);  // xg_b
  for (int s = 0; s < T_; ++s) {
    int tp = T_ - 1 - s;
    const float* Ap = (s == 0) ? zbuf : (bwd + (long)(tp + 1) * HH_);
    long lda = (s == 0) ? 512 : (long)T_ * HH_;
    gemm(Ap, lda, WhbT, 512, nullptr, R1 + (long)tp * 2048, (long)T_ * 2048,
         Gtmp, 2048, 64, 2048, 512);
    lstm_gate_kernel<<<dim3(128), dim3(256), 0, stream>>>(
        Gtmp, c_b, bwd + (long)tp * HH_, (long)T_ * HH_);
  }
  // --- GRU layer: xg_g = fwd@Wxg[0:512] + bwd@Wxg[512:] + bx_g -------------
  gemm(fwd, 512, WxgT, 1024, bx_g, nullptr, 0, R1, 3072, 8192, 3072, 512);
  gemm(bwd, 512, WxgT + 512, 1024, nullptr, R1, 3072, R1, 3072, 8192, 3072, 512);
  for (int t = 0; t < T_; ++t) {
    const float* Ap = (t == 0) ? zbuf : (gru + (long)(t - 1) * H_);
    long lda = (t == 0) ? 1024 : (long)T_ * H_;
    gemm(Ap, lda, WhgT, 1024, bh_g, nullptr, 0, Gtmp, 3072, 64, 3072, 1024);
    gru_gate_kernel<<<dim3(256), dim3(256), 0, stream>>>(
        R1 + (long)t * 3072, (long)T_ * 3072, Gtmp, Ap, lda,
        gru + (long)t * H_, (long)T_ * H_);
  }
  // --- block-GRU scan over t = 0..126 --------------------------------------
  const int acol0[BLK_] = {0, 0, 256, 512};     // lo_prev column offset
  const int wrow0[BLK_] = {0, 0, 0, 512};       // Wx_c row (K) offset
  const int Kb[BLK_]    = {256, 512, 512, 512}; // effective K
  for (int t = 0; t < T_ - 1; ++t) {
    for (int b = 0; b < BLK_; ++b) {
      gemm(gru + (long)t * H_ + acol0[b], (long)T_ * H_,
           WxcT + (long)b * 3072 * 1024 + wrow0[b], 1024,
           bx_c + (long)b * 3072, nullptr, 0,
           Gx + (long)b * 64 * 3072, 3072, 64, 3072, Kb[b]);
      gemm(hblk + (long)b * 64 * 1024, 1024,
           WhcT + (long)b * 3072 * 1024, 1024,
           bh_c + (long)b * 3072, nullptr, 0,
           Ghb + (long)b * 64 * 3072, 3072, 64, 3072, 1024);
    }
    blk_gate_kernel<<<dim3(256), dim3(256), 0, stream>>>(Gx, Ghb, hblk, wtb, zall, o_curr, t);
    wt_softmax_kernel<<<dim3(64), dim3(128), 0, stream>>>(zall, Ww, bw, wtb, o_wt, t);
  }
  // --- posterior / prior projections + outputs -----------------------------
  gemm(gru, 1024, WpT, 1024, bp, nullptr, 0, Ppost, 512, 8192, 512, 1024);
  post_split_kernel<<<dim3(8192), dim3(256), 0, stream>>>(Ppost, eps, o_pm, o_ps, o_z);
  gemm(o_curr, 1024, WprT, 1024, bpr, nullptr, 0, Pc, 512, 8128, 512, 1024);
  gemm(gru, (long)T_ * H_, WprT, 1024, bpr, nullptr, 0, P0, 512, 64, 512, 1024);
  prior_split_kernel<<<dim3(8192), dim3(256), 0, stream>>>(Pc, P0, o_prm, o_prs);
  copy_last_kernel<<<dim3(32512), dim3(256), 0, stream>>>(gru, o_last);
}
